// MSDeform_attn_transformer_encoder_only_74122545594643
// MI455X (gfx1250) — matrix operand order repaired and checked by reading, not compiled, not run on hardware
//
#include <hip/hip_runtime.h>

#define D 256
#define NH 8
#define HD 32
#define NP 4
#define NLVL 3
#define NLAYER 6
#define DFFN 1024
#define NTOK 13125
#define BATCH 8
#define MTOT (BATCH * NTOK) /* 105000 */
#define OFW 320             /* q-projection output stride: 288 padded to 64-multiple */

typedef __attribute__((ext_vector_type(16))) __bf16 bf16x16;
typedef __attribute__((ext_vector_type(8))) float floatx8;
typedef __attribute__((ext_vector_type(4))) unsigned int u32x4;
typedef __attribute__((ext_vector_type(4))) int i32x4;

#define AS1 __attribute__((address_space(1)))
#define AS3 __attribute__((address_space(3)))

#if __has_builtin(__builtin_amdgcn_global_load_async_to_lds_b128)
#define ASYNC_COPY 1
#else
#define ASYNC_COPY 0
#endif

union FragU {
    u32x4 u[2];
    bf16x16 v;
};

__device__ __forceinline__ unsigned short f2bf(float f) {
    unsigned int u = __float_as_uint(f);
    u += 0x7fffu + ((u >> 16) & 1u); // RNE
    return (unsigned short)(u >> 16);
}

__device__ __forceinline__ void wait_async0() {
#if __has_builtin(__builtin_amdgcn_s_wait_asynccnt)
    __builtin_amdgcn_s_wait_asynccnt(0);
#else
    asm volatile("s_wait_asynccnt 0x0" ::: "memory");
#endif
}

__device__ __forceinline__ void decode_tok(int n, int& lvl, int& r, int& c,
                                           int& hs, int& ws, int& start) {
    if (n < 10000) { lvl = 0; hs = 100; ws = 100; start = 0; r = n / 100; c = n - r * 100; }
    else if (n < 12500) { lvl = 1; hs = 50; ws = 50; start = 10000; int m = n - 10000; r = m / 50; c = m - r * 50; }
    else { lvl = 2; hs = 25; ws = 25; start = 12500; int m = n - 12500; r = m / 25; c = m - r * 25; }
}

// ---------------------------------------------------------------------------
// GEMM: C[M,Nout] = A[M,K](bf16) @ Wt[Nout,K](bf16, pre-transposed) + bias
// EPI: 0 -> f32 out, 1 -> relu + bf16 out, 2 -> bf16 out
// Block tile 128x64, 8 wave32 waves, each wave 32x32 (2x2 WMMA 16x16x32 bf16).
// Nout must be a multiple of 64 (weights are padded); M may be ragged.
// LDS tiles double-buffered; staged with GLOBAL_LOAD_ASYNC_TO_LDS_B128
// (ASYNCcnt-tracked DMA overlapped with WMMA) when the builtin exists.
// ---------------------------------------------------------------------------
template <int EPI>
__global__ __launch_bounds__(256) void gemm_bf16_wmma(
    const unsigned short* __restrict__ A, const unsigned short* __restrict__ Wt,
    const float* __restrict__ bias, void* __restrict__ outp, int M, int K, int Nout) {
#if ASYNC_COPY
    __shared__ __attribute__((aligned(16))) unsigned short sA[2][128 * 32];  // [row][k]
    __shared__ __attribute__((aligned(16))) unsigned short sBt[2][64 * 32];  // [n][k]
#else
    __shared__ __attribute__((aligned(16))) unsigned short sA[1][128 * 32];
    __shared__ __attribute__((aligned(16))) unsigned short sBt[1][64 * 32];
#endif

    const int t = threadIdx.x;
    const int m0 = blockIdx.x * 128;
    const int n0 = blockIdx.y * 64;
    const int lane = t & 31, wid = t >> 5;
    const int wm = wid & 3, wn = wid >> 2; // wave grid 4 (M) x 2 (N)
    const int lr = lane & 15, kh = lane >> 4;

    const int ar = t >> 1, ah = t & 1;  // A loader: 128 rows x 2 halves (32B each)
    const int br = t >> 2, bq4 = t & 3; // B loader: 64 rows x 4 chunks of 8 (16B each)
    const int arow = m0 + ar;

    floatx8 acc00 = {}, acc01 = {}, acc10 = {}, acc11 = {};

#if ASYNC_COPY
    // Issue one tile's global->LDS async DMA (per-lane 16B transfers).
    auto issue_tile = [&](int buf, int kk) {
        if (arow < M)
            __builtin_amdgcn_global_load_async_to_lds_b128(
                (AS1 i32x4*)(A + (size_t)arow * K + kk + ah * 16),
                (AS3 i32x4*)&sA[buf][ar * 32 + ah * 16], 0, 0);
        __builtin_amdgcn_global_load_async_to_lds_b128(
            (AS1 i32x4*)(Wt + (size_t)(n0 + br) * K + kk + bq4 * 8),
            (AS3 i32x4*)&sBt[buf][br * 32 + bq4 * 8], 0, 0);
    };
    issue_tile(0, 0);
    int buf = 0;
#endif

    for (int kk = 0; kk < K; kk += 32) {
#if ASYNC_COPY
        wait_async0();       // my DMA into sA/sBt[buf] done
        __syncthreads();     // everyone's DMA done, everyone done reading buf^1
        if (kk + 32 < K) issue_tile(buf ^ 1, kk + 32); // overlap next tile with WMMA
        const unsigned short* sAb = sA[buf];
        const unsigned short* sBb = sBt[buf];
#else
        __syncthreads();
        { // A tile
            u32x4 z0 = {0u, 0u, 0u, 0u}, z1 = {0u, 0u, 0u, 0u};
            if (arow < M) {
                const u32x4* pg = reinterpret_cast<const u32x4*>(A + (size_t)arow * K + kk + ah * 16);
                z0 = pg[0];
                z1 = pg[1];
            }
            u32x4* ps = reinterpret_cast<u32x4*>(&sA[0][ar * 32 + ah * 16]);
            ps[0] = z0;
            ps[1] = z1;
        }
        { // B tile (weights pre-transposed: contiguous along K)
            u32x4 bz = *reinterpret_cast<const u32x4*>(Wt + (size_t)(n0 + br) * K + kk + bq4 * 8);
            *reinterpret_cast<u32x4*>(&sBt[0][br * 32 + bq4 * 8]) = bz;
        }
        if (kk + 32 < K && arow < M)
            __builtin_prefetch(A + (size_t)arow * K + kk + 32, 0, 0);
        __syncthreads();
        const unsigned short* sAb = sA[0];
        const unsigned short* sBb = sBt[0];
#endif

        // Fragments (A 16-bit layout: lane-half K interleave 0-7/16-23 vs 8-15/24-31)
        FragU fa0, fa1, fb0, fb1;
        fa0.u[0] = *reinterpret_cast<const u32x4*>(&sAb[(wm * 32 + lr) * 32 + kh * 8]);
        fa0.u[1] = *reinterpret_cast<const u32x4*>(&sAb[(wm * 32 + lr) * 32 + 16 + kh * 8]);
        fa1.u[0] = *reinterpret_cast<const u32x4*>(&sAb[(wm * 32 + 16 + lr) * 32 + kh * 8]);
        fa1.u[1] = *reinterpret_cast<const u32x4*>(&sAb[(wm * 32 + 16 + lr) * 32 + 16 + kh * 8]);
        {
            const u32x4* p = reinterpret_cast<const u32x4*>(&sBb[(wn * 32 + lr) * 32 + kh * 8]);
            fb0.u[0] = p[0];
            fb0.u[1] = p[2];
            const u32x4* p1 = reinterpret_cast<const u32x4*>(&sBb[(wn * 32 + 16 + lr) * 32 + kh * 8]);
            fb1.u[0] = p1[0];
            fb1.u[1] = p1[2];
        }
        acc00 = __builtin_amdgcn_wmma_f32_16x16x32_bf16(false, fa0.v, false, fb0.v, (short)0, acc00, false, false);
        acc01 = __builtin_amdgcn_wmma_f32_16x16x32_bf16(false, fa0.v, false, fb1.v, (short)0, acc01, false, false);
        acc10 = __builtin_amdgcn_wmma_f32_16x16x32_bf16(false, fa1.v, false, fb0.v, (short)0, acc10, false, false);
        acc11 = __builtin_amdgcn_wmma_f32_16x16x32_bf16(false, fa1.v, false, fb1.v, (short)0, acc11, false, false);
#if ASYNC_COPY
        buf ^= 1;
#endif
    }

    float* outf = (float*)outp;
    unsigned short* outb = (unsigned short*)outp;
#pragma unroll
    for (int fm = 0; fm < 2; ++fm) {
#pragma unroll
        for (int fn = 0; fn < 2; ++fn) {
            floatx8 accv = (fm == 0) ? ((fn == 0) ? acc00 : acc01) : ((fn == 0) ? acc10 : acc11);
            int col = n0 + wn * 32 + fn * 16 + lr; // always < Nout (Nout % 64 == 0)
            float bcol = bias[col];
#pragma unroll
            for (int r = 0; r < 8; ++r) {
                int row = m0 + wm * 32 + fm * 16 + kh * 8 + r;
                if (row < M) {
                    float v = accv[r] + bcol;
                    if (EPI == 1) {
                        v = v > 0.f ? v : 0.f;
                        outb[(size_t)row * Nout + col] = f2bf(v);
                    } else if (EPI == 2) {
                        outb[(size_t)row * Nout + col] = f2bf(v);
                    } else {
                        outf[(size_t)row * Nout + col] = v;
                    }
                }
            }
        }
    }
}

// ---------------------------------------------------------------------------
// Flatten [B,D,H,W] levels -> x_f32 [B, N, D]
// ---------------------------------------------------------------------------
__global__ void prep_x_kernel(const float* __restrict__ x0, const float* __restrict__ x1,
                              const float* __restrict__ x2, float* __restrict__ xout) {
    size_t idx = (size_t)blockIdx.x * 256 + threadIdx.x;
    if (idx >= (size_t)MTOT * D) return;
    int d = (int)(idx & 255);
    int g = (int)(idx >> 8);
    int b = g / NTOK, n = g - b * NTOK;
    int lvl, r, c, hs, ws, st;
    decode_tok(n, lvl, r, c, hs, ws, st);
    const float* src = (lvl == 0) ? x0 : ((lvl == 1) ? x1 : x2);
    xout[idx] = src[(((size_t)b * D + d) * hs + r) * ws + c];
}

// q = bf16(x + pos + level_embed), xb = bf16(x)
__global__ void make_q_kernel(const float* __restrict__ xf, const float* __restrict__ p0,
                              const float* __restrict__ p1, const float* __restrict__ p2,
                              const float* __restrict__ le, unsigned short* __restrict__ qb,
                              unsigned short* __restrict__ xb) {
    size_t idx = (size_t)blockIdx.x * 256 + threadIdx.x;
    if (idx >= (size_t)MTOT * D) return;
    int d = (int)(idx & 255);
    int g = (int)(idx >> 8);
    int b = g / NTOK, n = g - b * NTOK;
    int lvl, r, c, hs, ws, st;
    decode_tok(n, lvl, r, c, hs, ws, st);
    const float* src = (lvl == 0) ? p0 : ((lvl == 1) ? p1 : p2);
    float pos = src[(((size_t)b * D + d) * hs + r) * ws + c] + le[lvl * D + d];
    float xv = xf[idx];
    qb[idx] = f2bf(xv + pos);
    xb[idx] = f2bf(xv);
}

// ---------------------------------------------------------------------------
// Deformable sampling: one wave per (token, head); lane = head-dim channel.
// offaw row stride OFW: [0,192) = offsets (h,l,p,2); [192,288) = logits (h, l*4+p)
// ---------------------------------------------------------------------------
__global__ __launch_bounds__(256) void msd_sample_kernel(const float* __restrict__ offaw,
                                                         const float* __restrict__ val,
                                                         unsigned short* __restrict__ attnb) {
    int wv = (int)((blockIdx.x * 256 + threadIdx.x) >> 5);
    int lane = threadIdx.x & 31;
    if (wv >= MTOT * NH) return;
    int g = wv >> 3, h = wv & 7;
    int b = g / NTOK, n = g - b * NTOK;
    int lvl, r, c, hs, ws, st;
    decode_tok(n, lvl, r, c, hs, ws, st);
    float refx = (c + 0.5f) / ws, refy = (r + 0.5f) / hs;
    const float* row = offaw + (size_t)g * OFW;

    float aw[12];
    float mx = -1e30f;
#pragma unroll
    for (int i = 0; i < 12; ++i) {
        aw[i] = row[192 + h * 12 + i];
        mx = fmaxf(mx, aw[i]);
    }
    float ssum = 0.f;
#pragma unroll
    for (int i = 0; i < 12; ++i) {
        aw[i] = __expf(aw[i] - mx);
        ssum += aw[i];
    }
    float inv = 1.f / ssum;

    float acc = 0.f;
    int start = 0;
    const int LS[3] = {100, 50, 25};
#pragma unroll
    for (int l = 0; l < 3; ++l) {
        int hl = LS[l], wl = LS[l];
#pragma unroll
        for (int p = 0; p < 4; ++p) {
            float ox = row[h * 24 + l * 8 + p * 2 + 0];
            float oy = row[h * 24 + l * 8 + p * 2 + 1];
            float fx = (refx + ox / (float)wl) * wl - 0.5f;
            float fy = (refy + oy / (float)hl) * hl - 0.5f;
            float x0f = floorf(fx), y0f = floorf(fy);
            float dx = fx - x0f, dy = fy - y0f;
            int ix0 = (int)x0f, iy0 = (int)y0f;
            float wgt = aw[l * 4 + p] * inv;
            float w00 = (1.f - dx) * (1.f - dy), w10 = dx * (1.f - dy);
            float w01 = (1.f - dx) * dy, w11 = dx * dy;
            const float* vb = val + ((size_t)(b * NTOK + start)) * D + h * HD + lane;
            if (ix0 >= 0 && ix0 < wl && iy0 >= 0 && iy0 < hl)
                acc += wgt * w00 * vb[(size_t)(iy0 * wl + ix0) * D];
            if (ix0 + 1 >= 0 && ix0 + 1 < wl && iy0 >= 0 && iy0 < hl)
                acc += wgt * w10 * vb[(size_t)(iy0 * wl + ix0 + 1) * D];
            if (ix0 >= 0 && ix0 < wl && iy0 + 1 >= 0 && iy0 + 1 < hl)
                acc += wgt * w01 * vb[(size_t)((iy0 + 1) * wl + ix0) * D];
            if (ix0 + 1 >= 0 && ix0 + 1 < wl && iy0 + 1 >= 0 && iy0 + 1 < hl)
                acc += wgt * w11 * vb[(size_t)((iy0 + 1) * wl + ix0 + 1) * D];
        }
        start += hl * hl;
    }
    attnb[(size_t)g * D + h * HD + lane] = f2bf(acc);
}

// ---------------------------------------------------------------------------
// x = LN(x + add) ; also write bf16 copy. One block (256 thr) per token.
// ---------------------------------------------------------------------------
__global__ __launch_bounds__(256) void ln_kernel(float* __restrict__ x,
                                                 const float* __restrict__ addsrc,
                                                 const float* __restrict__ g,
                                                 const float* __restrict__ b,
                                                 unsigned short* __restrict__ xb) {
    __shared__ float red[256];
    int t = threadIdx.x;
    size_t base = (size_t)blockIdx.x * D;
    float v = x[base + t] + addsrc[base + t];
    red[t] = v;
    __syncthreads();
    for (int s = 128; s > 0; s >>= 1) {
        if (t < s) red[t] += red[t + s];
        __syncthreads();
    }
    float mu = red[0] * (1.f / D);
    __syncthreads();
    red[t] = v * v;
    __syncthreads();
    for (int s = 128; s > 0; s >>= 1) {
        if (t < s) red[t] += red[t + s];
        __syncthreads();
    }
    float var = red[0] * (1.f / D) - mu * mu;
    float y = (v - mu) * rsqrtf(var + 1e-5f) * g[t] + b[t];
    x[base + t] = y;
    xb[base + t] = f2bf(y);
}

// ---------------------------------------------------------------------------
// Weight preparation: bf16 + transpose (W[K,N] -> Wt[N,K]) so GEMM B-tiles are
// contiguous 16B chunks (async-DMA-friendly). q-projection is concatenated
// (W_off | W_attw | 0-pad) to OFW=320 columns.
// ---------------------------------------------------------------------------
__global__ void prep_wq_kernel(const float* __restrict__ Woff, const float* __restrict__ Watt,
                               unsigned short* __restrict__ Wqt) {
    int idx = blockIdx.x * 256 + threadIdx.x;
    if (idx >= NLAYER * OFW * D) return;
    int k = idx % D;
    int rest = idx / D;
    int col = rest % OFW;
    int li = rest / OFW;
    float v = 0.f;
    if (col < 192) v = Woff[((size_t)li * D + k) * 192 + col];
    else if (col < 288) v = Watt[((size_t)li * D + k) * 96 + (col - 192)];
    Wqt[idx] = f2bf(v);
}

__global__ void prep_bq_kernel(const float* __restrict__ boff, const float* __restrict__ batt,
                               float* __restrict__ bq) {
    int idx = blockIdx.x * 256 + threadIdx.x;
    if (idx >= NLAYER * OFW) return;
    int col = idx % OFW, li = idx / OFW;
    float v = 0.f;
    if (col < 192) v = boff[li * 192 + col];
    else if (col < 288) v = batt[li * 96 + (col - 192)];
    bq[idx] = v;
}

// src [NLAYER][K][N] f32 -> dst [NLAYER][N][K] bf16
__global__ void conv_t_bf16_kernel(const float* __restrict__ src, unsigned short* __restrict__ dst,
                                   int K, int N) {
    size_t idx = (size_t)blockIdx.x * 256 + threadIdx.x;
    size_t per = (size_t)K * N;
    if (idx >= (size_t)NLAYER * per) return;
    int li = (int)(idx / per);
    size_t rem = idx - (size_t)li * per;
    int n = (int)(rem / K);
    int k = (int)(rem - (size_t)n * K);
    dst[idx] = f2bf(src[(size_t)li * per + (size_t)k * N + n]);
}

__global__ void write_out_kernel(const float* __restrict__ x, float* __restrict__ out, int total) {
    int idx = blockIdx.x * 256 + threadIdx.x;
    if (idx < total) out[idx] = x[idx];
    if (idx < 6) {
        const int sh0[6] = {100, 100, 50, 50, 25, 25};
        ((int*)out)[total + idx] = sh0[idx];
    }
}

// ---------------------------------------------------------------------------
extern "C" void kernel_launch(void* const* d_in, const int* in_sizes, int n_in, void* d_out,
                              int out_size, void* d_ws, size_t ws_size, hipStream_t stream) {
    (void)in_sizes; (void)n_in; (void)out_size; (void)ws_size;
    const float* x0 = (const float*)d_in[0];
    const float* p0 = (const float*)d_in[1];
    const float* x1 = (const float*)d_in[2];
    const float* p1 = (const float*)d_in[3];
    const float* x2 = (const float*)d_in[4];
    const float* p2 = (const float*)d_in[5];
    const float* le = (const float*)d_in[6];
    const float* Woff = (const float*)d_in[7];
    const float* boff = (const float*)d_in[8];
    const float* Watt = (const float*)d_in[9];
    const float* batt = (const float*)d_in[10];
    const float* Wval = (const float*)d_in[11];
    const float* bval = (const float*)d_in[12];
    const float* Wout = (const float*)d_in[13];
    const float* bout = (const float*)d_in[14];
    const float* ln1g = (const float*)d_in[15];
    const float* ln1b = (const float*)d_in[16];
    const float* Wff1 = (const float*)d_in[17];
    const float* bff1 = (const float*)d_in[18];
    const float* Wff2 = (const float*)d_in[19];
    const float* bff2 = (const float*)d_in[20];
    const float* ln2g = (const float*)d_in[21];
    const float* ln2b = (const float*)d_in[22];

    char* ws = (char*)d_ws;
    size_t off = 0;
    auto alloc = [&](size_t bytes) -> char* {
        char* p = ws + off;
        off += (bytes + 255) & ~(size_t)255;
        return p;
    };

    float* x_f32 = (float*)alloc((size_t)MTOT * D * 4);
    unsigned short* q_bf = (unsigned short*)alloc((size_t)MTOT * D * 2);
    unsigned short* x_bf = (unsigned short*)alloc((size_t)MTOT * D * 2);
    float* val_f32 = (float*)alloc((size_t)MTOT * D * 4);
    float* offaw = (float*)alloc((size_t)MTOT * OFW * 4);
    float* tmp = offaw; // aliased: offaw fully consumed before tmp is produced
    unsigned short* attn_bf = (unsigned short*)alloc((size_t)MTOT * D * 2);
    unsigned short* h_bf = (unsigned short*)alloc((size_t)MTOT * DFFN * 2);
    unsigned short* Wqt_bf = (unsigned short*)alloc((size_t)NLAYER * OFW * D * 2);
    float* bq = (float*)alloc((size_t)NLAYER * OFW * 4);
    unsigned short* Wvalt_bf = (unsigned short*)alloc((size_t)NLAYER * D * D * 2);
    unsigned short* Woutt_bf = (unsigned short*)alloc((size_t)NLAYER * D * D * 2);
    unsigned short* Wff1t_bf = (unsigned short*)alloc((size_t)NLAYER * D * DFFN * 2);
    unsigned short* Wff2t_bf = (unsigned short*)alloc((size_t)NLAYER * DFFN * D * 2);

    // One-time weight prep (bf16 + transpose + q-projection concat/pad)
    {
        int tq = NLAYER * OFW * D;
        prep_wq_kernel<<<(tq + 255) / 256, 256, 0, stream>>>(Woff, Watt, Wqt_bf);
        int tb = NLAYER * OFW;
        prep_bq_kernel<<<(tb + 255) / 256, 256, 0, stream>>>(boff, batt, bq);
        size_t n1 = (size_t)NLAYER * D * D;
        conv_t_bf16_kernel<<<(int)((n1 + 255) / 256), 256, 0, stream>>>(Wval, Wvalt_bf, D, D);
        conv_t_bf16_kernel<<<(int)((n1 + 255) / 256), 256, 0, stream>>>(Wout, Woutt_bf, D, D);
        size_t n2 = (size_t)NLAYER * D * DFFN;
        conv_t_bf16_kernel<<<(int)((n2 + 255) / 256), 256, 0, stream>>>(Wff1, Wff1t_bf, D, DFFN);
        conv_t_bf16_kernel<<<(int)((n2 + 255) / 256), 256, 0, stream>>>(Wff2, Wff2t_bf, DFFN, D);
    }

    size_t tot_xd = (size_t)MTOT * D;
    int nb_xd = (int)((tot_xd + 255) / 256);
    prep_x_kernel<<<nb_xd, 256, 0, stream>>>(x0, x1, x2, x_f32);

    dim3 blk(256);
    int gm = (MTOT + 127) / 128;
    for (int li = 0; li < NLAYER; ++li) {
        make_q_kernel<<<nb_xd, 256, 0, stream>>>(x_f32, p0, p1, p2, le, q_bf, x_bf);
        // offsets + attention logits: offaw = q @ Wq + bq   [M, OFW]
        gemm_bf16_wmma<0><<<dim3(gm, OFW / 64), blk, 0, stream>>>(
            q_bf, Wqt_bf + (size_t)li * OFW * D, bq + li * OFW, offaw, MTOT, D, OFW);
        // values: val = x @ Wval + bval   [M, 256]
        gemm_bf16_wmma<0><<<dim3(gm, D / 64), blk, 0, stream>>>(
            x_bf, Wvalt_bf + (size_t)li * D * D, bval + li * D, val_f32, MTOT, D, D);
        // deformable sampling -> attn_bf [M, 256]
        msd_sample_kernel<<<(MTOT * NH * 32 + 255) / 256, 256, 0, stream>>>(offaw, val_f32, attn_bf);
        // output projection: tmp = attn @ Wout + bout
        gemm_bf16_wmma<0><<<dim3(gm, D / 64), blk, 0, stream>>>(
            attn_bf, Woutt_bf + (size_t)li * D * D, bout + li * D, tmp, MTOT, D, D);
        // x = LN(x + attn_out)
        ln_kernel<<<MTOT, 256, 0, stream>>>(x_f32, tmp, ln1g + li * D, ln1b + li * D, x_bf);
        // FFN: h = relu(x @ Wff1 + b)  (bf16 out)
        gemm_bf16_wmma<1><<<dim3(gm, DFFN / 64), blk, 0, stream>>>(
            x_bf, Wff1t_bf + (size_t)li * D * DFFN, bff1 + li * DFFN, h_bf, MTOT, D, DFFN);
        // tmp = h @ Wff2 + b
        gemm_bf16_wmma<0><<<dim3(gm, D / 64), blk, 0, stream>>>(
            h_bf, Wff2t_bf + (size_t)li * DFFN * D, bff2 + li * D, tmp, MTOT, DFFN, D);
        // x = LN(x + ffn)
        ln_kernel<<<MTOT, 256, 0, stream>>>(x_f32, tmp, ln2g + li * D, ln2b + li * D, x_bf);
    }
    write_out_kernel<<<nb_xd, 256, 0, stream>>>(x_f32, (float*)d_out, (int)tot_xd);
}
